// EfficientAttention_3040836846418
// MI455X (gfx1250) — matrix operand order repaired and checked by reading, not compiled, not run on hardware
//
#include <hip/hip_runtime.h>

typedef __attribute__((ext_vector_type(16))) _Float16 v16h;
typedef __attribute__((ext_vector_type(8)))  float    v8f;

#define EA_B 32
#define EA_N 16384
#define EA_D 64

// ---------------------------------------------------------------------------
// Pass 1: ctx partials.  grid = (16, B), block = 128 threads (4 waves).
// Wave w of WG wg handles rows [wg*1024 + w*256, +256) of batch blockIdx.y,
// in 8 K-steps of 32 rows.  ctx_tile[d][e] += sum_n softmax(k)[n][d] * v[n][e]
// WMMA view per K-step: A(16x32) = k_n^T tile (M=d, K=n), B(32x16) = v tile.
// ---------------------------------------------------------------------------
__global__ __launch_bounds__(128)
void ea_ctx_kernel(const float* __restrict__ K, const float* __restrict__ V,
                   float* __restrict__ partials) {
  __shared__ _Float16 kbuf[4][32][64];   // 16 KB
  __shared__ _Float16 vbuf[4][32][64];   // 16 KB
  __shared__ float    ctile[64][64];     // 16 KB

  const int tid  = threadIdx.x;
  const int wave = tid >> 5;
  const int lane = tid & 31;
  const int m16  = lane & 15;
  const int hi   = lane >> 4;            // 0 = lanes 0-15, 1 = lanes 16-31
  const int b    = blockIdx.y;
  const int wg   = blockIdx.x;

  // zero the shared accumulation tile
  float* ctf = &ctile[0][0];
#pragma unroll
  for (int i = 0; i < 32; ++i) ctf[tid + i * 128] = 0.0f;
  __syncthreads();

  const long base     = (long)b * EA_N * EA_D;
  const int  row_base = wg * 1024 + wave * 256;

  v8f acc[4][4];
#pragma unroll
  for (int i = 0; i < 4; ++i)
#pragma unroll
    for (int j = 0; j < 4; ++j)
      acc[i][j] = v8f{0.f, 0.f, 0.f, 0.f, 0.f, 0.f, 0.f, 0.f};

  for (int step = 0; step < 8; ++step) {
    const int n0 = row_base + step * 32;
    const float4* krow = (const float4*)(K + base + (long)(n0 + lane) * EA_D);
    const float4* vrow = (const float4*)(V + base + (long)(n0 + lane) * EA_D);

    // per-lane fp32 softmax of one 64-wide k row
    float r[64];
    float mx = -3.0e38f;
#pragma unroll
    for (int i = 0; i < 16; ++i) {
      float4 t = krow[i];
      r[4*i+0] = t.x; r[4*i+1] = t.y; r[4*i+2] = t.z; r[4*i+3] = t.w;
      mx = fmaxf(mx, fmaxf(fmaxf(t.x, t.y), fmaxf(t.z, t.w)));
    }
    float s = 0.f;
#pragma unroll
    for (int j = 0; j < 64; ++j) { float e = __expf(r[j] - mx); r[j] = e; s += e; }
    const float inv = 1.0f / s;
#pragma unroll
    for (int j = 0; j < 64; ++j) kbuf[wave][lane][j] = (_Float16)(r[j] * inv);

    // stage v row as f16
#pragma unroll
    for (int i = 0; i < 16; ++i) {
      float4 t = vrow[i];
      vbuf[wave][lane][4*i+0] = (_Float16)t.x;
      vbuf[wave][lane][4*i+1] = (_Float16)t.y;
      vbuf[wave][lane][4*i+2] = (_Float16)t.z;
      vbuf[wave][lane][4*i+3] = (_Float16)t.w;
    }
    __syncthreads();

    // A fragments: 16-bit A 16x32 layout (contiguous K order);
    // half j holds K = j + 16*hi, row M = lane&15.
    v16h afrag[4];
#pragma unroll
    for (int dt = 0; dt < 4; ++dt) {
      v16h a;
#pragma unroll
      for (int j = 0; j < 16; ++j) {
        const int kk = j + hi * 16;                     // n within 32-row chunk
        a[j] = kbuf[wave][kk][dt * 16 + m16];          // k_n[n][d] -> A[d][n]
      }
      afrag[dt] = a;
    }
    // B fragments: 16-bit B 32x16 layout. half j holds K = j + 16*hi, N = lane&15.
    v16h bfrag[4];
#pragma unroll
    for (int et = 0; et < 4; ++et) {
      v16h bb;
#pragma unroll
      for (int j = 0; j < 16; ++j) {
        const int kk = j + hi * 16;
        bb[j] = vbuf[wave][kk][et * 16 + m16];         // v[n][e] -> B[n][e]
      }
      bfrag[et] = bb;
    }

#pragma unroll
    for (int dt = 0; dt < 4; ++dt)
#pragma unroll
      for (int et = 0; et < 4; ++et)
        acc[dt][et] = __builtin_amdgcn_wmma_f32_16x16x32_f16(
            false, afrag[dt], false, bfrag[et], (short)0, acc[dt][et],
            false, false);
    __syncthreads();
  }

  // deterministic merge of the 4 wave tiles into ctile (sequenced by wave id)
#pragma unroll
  for (int w = 0; w < 4; ++w) {
    if (wave == w) {
#pragma unroll
      for (int dt = 0; dt < 4; ++dt)
#pragma unroll
        for (int et = 0; et < 4; ++et)
#pragma unroll
          for (int r8 = 0; r8 < 8; ++r8)
            ctile[dt * 16 + r8 + 8 * hi][et * 16 + m16] += acc[dt][et][r8];
    }
    __syncthreads();
  }

  // write this WG's 64x64 partial tile (coalesced)
  float* dst = partials + (long)(b * 16 + wg) * 4096;
#pragma unroll
  for (int i = 0; i < 32; ++i) dst[tid + i * 128] = ctf[tid + i * 128];
}

// ---------------------------------------------------------------------------
// Pass 2: reduce 16 partial tiles per batch into the final ctx (L2-resident).
// ---------------------------------------------------------------------------
__global__ __launch_bounds__(256)
void ea_reduce_kernel(const float* __restrict__ partials, float* __restrict__ ctx) {
  const int idx = blockIdx.x * 256 + threadIdx.x;   // 0 .. 131071
  const int b = idx >> 12;
  const int e = idx & 4095;
  const float* p = partials + (long)b * 16 * 4096 + e;
  float s = 0.f;
#pragma unroll
  for (int w = 0; w < 16; ++w) s += p[w * 4096];
  ctx[idx] = s;
}

// ---------------------------------------------------------------------------
// Pass 3: out = softmax(q) @ ctx.   grid = (64, B), block = 32 (one wave).
// Each wave streams 256 rows (8 steps of 32 rows); ctx loaded once as 8
// B fragments (K=64 -> 2 WMMA K-steps).
// ---------------------------------------------------------------------------
__global__ __launch_bounds__(32)
void ea_out_kernel(const float* __restrict__ Q, const float* __restrict__ ctx,
                   float* __restrict__ out) {
  __shared__ _Float16 qbuf[32][64];
  const int lane = threadIdx.x;
  const int m16  = lane & 15;
  const int hi   = lane >> 4;
  const int b    = blockIdx.y;
  const long base = (long)b * EA_N * EA_D;
  const float* cb = ctx + b * 4096;

  // B fragments from ctx: B[Kl][e] = ctx[ks*32 + Kl][e]; half j -> Kl = j + 16*hi
  v16h bfrag[2][4];
#pragma unroll
  for (int ks = 0; ks < 2; ++ks)
#pragma unroll
    for (int et = 0; et < 4; ++et) {
      v16h bb;
#pragma unroll
      for (int j = 0; j < 16; ++j)
        bb[j] = (_Float16)cb[(ks * 32 + j + 16 * hi) * 64 + et * 16 + m16];
      bfrag[ks][et] = bb;
    }

  const int row0 = blockIdx.x * 256;
  for (int it = 0; it < 8; ++it) {
    const int n0 = row0 + it * 32;
    const float4* qrow = (const float4*)(Q + base + (long)(n0 + lane) * EA_D);

    float r[64];
    float mx = -3.0e38f;
#pragma unroll
    for (int i = 0; i < 16; ++i) {
      float4 t = qrow[i];
      r[4*i+0] = t.x; r[4*i+1] = t.y; r[4*i+2] = t.z; r[4*i+3] = t.w;
      mx = fmaxf(mx, fmaxf(fmaxf(t.x, t.y), fmaxf(t.z, t.w)));
    }
    float s = 0.f;
#pragma unroll
    for (int j = 0; j < 64; ++j) { float e = __expf(r[j] - mx); r[j] = e; s += e; }
    const float inv = 1.0f / s;
#pragma unroll
    for (int j = 0; j < 64; ++j) qbuf[lane][j] = (_Float16)(r[j] * inv);
    __syncthreads();

    // A fragments: A[M][Kl] = q_n[n0 + mt*16 + M][ks*32 + Kl]
    v16h afrag[2][2];
#pragma unroll
    for (int mt = 0; mt < 2; ++mt)
#pragma unroll
      for (int ks = 0; ks < 2; ++ks) {
        v16h a;
#pragma unroll
        for (int j = 0; j < 16; ++j) {
          const int kk = j + hi * 16;
          a[j] = qbuf[mt * 16 + m16][ks * 32 + kk];
        }
        afrag[mt][ks] = a;
      }

#pragma unroll
    for (int mt = 0; mt < 2; ++mt)
#pragma unroll
      for (int et = 0; et < 4; ++et) {
        v8f acc = v8f{0.f, 0.f, 0.f, 0.f, 0.f, 0.f, 0.f, 0.f};
        acc = __builtin_amdgcn_wmma_f32_16x16x32_f16(
            false, afrag[mt][0], false, bfrag[0][et], (short)0, acc, false, false);
        acc = __builtin_amdgcn_wmma_f32_16x16x32_f16(
            false, afrag[mt][1], false, bfrag[1][et], (short)0, acc, false, false);
#pragma unroll
        for (int r8 = 0; r8 < 8; ++r8)
          out[base + (long)(n0 + mt * 16 + r8 + 8 * hi) * 64 + et * 16 + m16] =
              acc[r8];
      }
    __syncthreads();
  }
}

// ---------------------------------------------------------------------------
extern "C" void kernel_launch(void* const* d_in, const int* in_sizes, int n_in,
                              void* d_out, int out_size, void* d_ws, size_t ws_size,
                              hipStream_t stream) {
  (void)in_sizes; (void)n_in; (void)out_size; (void)ws_size;
  const float* q = (const float*)d_in[0];
  const float* k = (const float*)d_in[1];
  const float* v = (const float*)d_in[2];
  float* out = (float*)d_out;

  float* partials = (float*)d_ws;                   // 512 tiles * 4096 f32 = 8 MB
  float* ctx      = partials + (long)512 * 4096;    // 131072 f32 = 512 KB

  ea_ctx_kernel<<<dim3(16, EA_B), 128, 0, stream>>>(k, v, partials);
  ea_reduce_kernel<<<dim3(131072 / 256), 256, 0, stream>>>(partials, ctx);
  ea_out_kernel<<<dim3(64, EA_B), 32, 0, stream>>>(q, ctx, out);
}
